// FastConvLSTM_32530082300216
// MI455X (gfx1250) — 16-bit and 8-bit matrix operand order repaired and checked by reading, compiled, not run on hardware
//
#include <hip/hip_runtime.h>

typedef __attribute__((ext_vector_type(16))) _Float16 v16h;
typedef __attribute__((ext_vector_type(8)))  float    v8f;
typedef __attribute__((ext_vector_type(8)))  int      v8i;

#define WS_W1F    0
#define WS_W2F    16384
#define WS_BCP    147456
#define WS_WFRAG  163840
#define WS_XT     1048576

template <bool HI>
__device__ __forceinline__ unsigned fp8x2(float a, float b, unsigned old) {
  return (unsigned)__builtin_amdgcn_cvt_pk_fp8_f32(a, b, (int)old, HI);
}
__device__ __forceinline__ unsigned char to_fp8(float v) {
  return (unsigned char)(fp8x2<false>(v, 0.f, 0u) & 0xffu);
}
__device__ __forceinline__ float sigm(float x) { return 1.f / (1.f + __expf(-x)); }

__global__ void prep_enc1(const float* __restrict__ cw, const float* __restrict__ cb,
                          const float* __restrict__ gam, const float* __restrict__ bet,
                          const float* __restrict__ mu, const float* __restrict__ var,
                          _Float16* __restrict__ w1f, float* __restrict__ bcp) {
  int idx = blockIdx.x * 256 + threadIdx.x;
  if (idx < 64) {
    float s = gam[idx] * rsqrtf(var[idx] + 1e-5f);
    bcp[idx] = (cb[idx] - mu[idx]) * s + bet[idx];
  }
  if (idx < 6144) {
    int blk = idx >> 9, rem = idx & 511, lane = rem >> 4, hv = rem & 15;
    int ntl = blk / 3, k = blk % 3, v = hv >> 1, j = hv & 1;
    int c = ntl * 16 + (lane & 15);
    int f = k * 32 + (lane >> 4) * 16 + v * 2 + j;
    float val = 0.f;
    if (f < 84) {
      float s = gam[c] * rsqrtf(var[c] + 1e-5f);
      val = s * cw[(c * 84 + f) * 3 + 1];
    }
    w1f[idx] = (_Float16)val;
  }
}

__global__ void prep_enc2(const float* __restrict__ lw, _Float16* __restrict__ w2f) {
  int idx = blockIdx.x * 256 + threadIdx.x;
  if (idx >= 65536) return;
  int blk = idx >> 9, rem = idx & 511, lane = rem >> 4, hv = rem & 15;
  int ot = blk >> 1, k = blk & 1, v = hv >> 1, j = hv & 1;
  int o = ot * 16 + (lane & 15);
  int c = k * 32 + (lane >> 4) * 16 + v * 2 + j;
  float s = 0.f;
  const float* row = lw + (size_t)o * 1024 + c * 16;
  #pragma unroll
  for (int jj = 0; jj < 16; ++jj) s += row[jj];
  w2f[idx] = (_Float16)s;
}

__global__ void prep_cell(const float* __restrict__ cw, unsigned char* __restrict__ wf) {
  int idx = blockIdx.x * 256 + threadIdx.x;
  if (idx >= 229376) return;
  int blk = idx >> 10, rem = idx & 1023, lane = rem >> 5, bb = rem & 31;
  int k = blk % 7, mtg = blk / 7;
  int mt = mtg & 7, g = mtg >> 3;
  int v = bb >> 2, j = bb & 3;
  int K = k * 64 + (v >> 2) * 32 + (lane >> 4) * 16 + (v & 3) * 4 + j;
  float val = 0.f;
  if (K < 432) {
    int tau = K / 48, ci = K % 48;
    int o = g * 128 + mt * 16 + (lane & 15);
    val = cw[((size_t)(o * 48 + ci) * 3 + tau / 3) * 3 + (tau % 3)];
  }
  wf[idx] = to_fp8(val);
}

__global__ __launch_bounds__(256) void enc_kernel(
    const float* __restrict__ x, const _Float16* __restrict__ w1f,
    const _Float16* __restrict__ w2f, const float* __restrict__ bcp,
    const float* __restrict__ lin_b, unsigned char* __restrict__ xt) {
  extern __shared__ char sm[];
  const int W1F = 0, W2F = 12288, BC = 143360, LB = 143616, XTW = 147712, ENCB = 172288;
  int tid = threadIdx.x, lane = tid & 31, wave = tid >> 5;
  int hf = lane >> 4, p = lane & 15;

  for (int i = tid; i < 768;  i += 256) *(int4*)(sm + W1F + i * 16) = ((const int4*)w1f)[i];
  for (int i = tid; i < 8192; i += 256) *(int4*)(sm + W2F + i * 16) = ((const int4*)w2f)[i];
  if (tid < 64) *(float*)(sm + BC + tid * 4) = bcp[tid];
  for (int i = tid; i < 1024; i += 256) *(float*)(sm + LB + i * 4) = lin_b[i];
  __syncthreads();

  const int XTo = XTW + wave * 3072;
  const int ENo = ENCB + wave * 2048;
  const v8f zz = {0.f,0.f,0.f,0.f,0.f,0.f,0.f,0.f};

  for (int tile = blockIdx.x * 8 + wave; tile < 16384; tile += 2048) {
    for (int h = lane; h < 1536; h += 32) {
      int row = h / 96, f = h - row * 96;
      float val = (f < 84) ? x[(size_t)tile * 1344 + row * 84 + f] : 0.f;
      *(_Float16*)(sm + XTo + h * 2) = (_Float16)val;
    }
    #pragma unroll
    for (int ntl = 0; ntl < 4; ++ntl) {
      v8f a = zz;
      #pragma unroll
      for (int k = 0; k < 3; ++k) {
        union { int4 q[2]; v16h v; } A;
        A.q[0] = *(const int4*)(sm + XTo + p * 192 + k * 64 + hf * 32);
        A.q[1] = *(const int4*)(sm + XTo + p * 192 + k * 64 + hf * 32 + 16);
        v16h B = *(const v16h*)(sm + W1F + (ntl * 3 + k) * 1024 + lane * 32);
        a = __builtin_amdgcn_wmma_f32_16x16x32_f16(false, A.v, false, B, (short)0, a, false, false);
      }
      float bc = *(const float*)(sm + BC + (ntl * 16 + p) * 4);
      #pragma unroll
      for (int r = 0; r < 8; ++r) {
        float e = fmaxf(a[r] + bc, 0.f);
        *(_Float16*)(sm + ENo + (r + 8 * hf) * 128 + (ntl * 16 + p) * 2) = (_Float16)e;
      }
    }
    int grow = tile * 16 + p;
    size_t xb = (size_t)((grow & 127) * 2048 + (grow >> 7)) * 1024;
    for (int ot = 0; ot < 64; ++ot) {
      v8f a = zz;
      #pragma unroll
      for (int k = 0; k < 2; ++k) {
        v16h A = *(const v16h*)(sm + W2F + (ot * 2 + k) * 1024 + lane * 32);
        v16h B = *(const v16h*)(sm + ENo + p * 128 + k * 64 + hf * 32);
        a = __builtin_amdgcn_wmma_f32_16x16x32_f16(false, A, false, B, (short)0, a, false, false);
      }
      float4 b0 = *(const float4*)(sm + LB + (ot * 16 + hf * 8) * 4);
      float4 b1 = *(const float4*)(sm + LB + (ot * 16 + hf * 8) * 4 + 16);
      unsigned lo = fp8x2<false>(a[0] + b0.x, a[1] + b0.y, 0u);
      lo = fp8x2<true>(a[2] + b0.z, a[3] + b0.w, lo);
      unsigned hi = fp8x2<false>(a[4] + b1.x, a[5] + b1.y, 0u);
      hi = fp8x2<true>(a[6] + b1.z, a[7] + b1.w, hi);
      unsigned* dst = (unsigned*)(xt + xb + ot * 16 + hf * 8);
      dst[0] = lo; dst[1] = hi;
    }
  }
}

__global__ __launch_bounds__(256) void scan_kernel(
    const unsigned char* __restrict__ xt, const unsigned char* __restrict__ wfrag,
    const float* __restrict__ cell_b, const float* __restrict__ cls1_w,
    const float* __restrict__ cls1_b, const float* __restrict__ cls2_w,
    const float* __restrict__ cls2_b, float* __restrict__ out) {
  extern __shared__ char sm[];
  const int WF = 0, PAD = 229376, ZERO = 257024, BIAS = 257280;
  const int HB = 0, HID = 32768;
  int tid = threadIdx.x, lane = tid & 31, wave = tid >> 5;
  int hf = lane >> 4, p = lane & 15;
  int nt = wave & 3, mtb = (wave >> 2) * 4;

  for (int i = tid; i < 14336; i += 256) *(int4*)(sm + WF + i * 16) = ((const int4*)wfrag)[i];
  for (int i = tid; i < 6976;  i += 256) *(int*)(sm + PAD + i * 4) = 0;
  for (int i = tid; i < 512;   i += 256) *(float*)(sm + BIAS + i * 4) = cell_b[i];
  __syncthreads();

  int bs = tid >> 6, cs = tid & 63;
  int bg = blockIdx.x * 4 + bs;
  {
    int4 x0 = *(const int4*)(xt + (size_t)bg * 1024 + cs * 16);
    unsigned u[4] = {(unsigned)x0.x, (unsigned)x0.y, (unsigned)x0.z, (unsigned)x0.w};
    #pragma unroll
    for (int j = 0; j < 16; ++j)
      sm[PAD + bs * 6912 + ((j >> 2) + 1) * 1152 + ((j & 3) + 1) * 192 + cs] =
          (char)((u[j >> 2] >> (8 * (j & 3))) & 0xffu);
  }
  __syncthreads();

  const v8f zz = {0.f,0.f,0.f,0.f,0.f,0.f,0.f,0.f};
  v8f cst[4];
  #pragma unroll
  for (int m = 0; m < 4; ++m) cst[m] = zz;

  int boff[7][2];
  #pragma unroll
  for (int k = 0; k < 7; ++k) {
    #pragma unroll
    for (int q = 0; q < 2; ++q) {
      int m16 = k * 4 + q * 2 + hf;
      if (m16 < 27) {
        int tau = m16 / 3;
        int p6 = (p / 4 + tau / 3) * 6 + (p % 4 + tau % 3);
        boff[k][q] = PAD + nt * 6912 + p6 * 192 + (m16 % 3) * 16;
      } else boff[k][q] = ZERO;
    }
  }
  int hoff = PAD + nt * 6912 + ((p >> 2) + 1) * 1152 + ((p & 3) + 1) * 192 + 64;

  int4 xn;
  #pragma unroll 1
  for (int t = 0; t < 128; ++t) {
    if (t < 127)
      xn = *(const int4*)(xt + (size_t)((t + 1) * 2048 + bg) * 1024 + cs * 16);

    v8f acc[4][4];
    #pragma unroll
    for (int g = 0; g < 4; ++g)
      #pragma unroll
      for (int m = 0; m < 4; ++m) acc[g][m] = zz;

    #pragma unroll
    for (int k = 0; k < 7; ++k) {
      #pragma unroll
      for (int g = 0; g < 4; ++g) {
        int4 bl0 = *(const int4*)(sm + boff[k][0] + g * 48);
        int4 bl1 = *(const int4*)(sm + boff[k][1] + g * 48);
        v8i bfr = {bl0.x, bl0.y, bl0.z, bl0.w, bl1.x, bl1.y, bl1.z, bl1.w};
        #pragma unroll
        for (int m = 0; m < 4; ++m) {
          v8i afr = *(const v8i*)(sm + WF + ((g * 8 + (mtb + m)) * 7 + k) * 1024 + lane * 32);
          acc[g][m] = __builtin_amdgcn_wmma_f32_16x16x64_fp8_fp8(afr, bfr, (short)0,
                                                                 acc[g][m], false, false);
        }
      }
    }
    __syncthreads();

    #pragma unroll
    for (int m = 0; m < 4; ++m) {
      int mt = mtb + m;
      float bsv[4][8];
      #pragma unroll
      for (int g = 0; g < 4; ++g) {
        float4 u0 = *(const float4*)(sm + BIAS + (g * 128 + mt * 16 + hf * 8) * 4);
        float4 u1 = *(const float4*)(sm + BIAS + (g * 128 + mt * 16 + hf * 8) * 4 + 16);
        bsv[g][0] = u0.x; bsv[g][1] = u0.y; bsv[g][2] = u0.z; bsv[g][3] = u0.w;
        bsv[g][4] = u1.x; bsv[g][5] = u1.y; bsv[g][6] = u1.z; bsv[g][7] = u1.w;
      }
      float hv[8];
      #pragma unroll
      for (int r = 0; r < 8; ++r) {
        float iv = acc[0][m][r] + bsv[0][r];
        float fv = acc[1][m][r] + bsv[1][r];
        float ov = acc[2][m][r] + bsv[2][r];
        float gv = acc[3][m][r] + bsv[3][r];
        float cn = sigm(fv) * cst[m][r] + sigm(iv) * tanhf(gv);
        cst[m][r] = cn;
        hv[r] = sigm(ov) * tanhf(cn);
      }
      unsigned pk0 = fp8x2<false>(hv[0], hv[1], 0u); pk0 = fp8x2<true>(hv[2], hv[3], pk0);
      unsigned pk1 = fp8x2<false>(hv[4], hv[5], 0u); pk1 = fp8x2<true>(hv[6], hv[7], pk1);
      *(unsigned*)(sm + hoff + mt * 16 + hf * 8)     = pk0;
      *(unsigned*)(sm + hoff + mt * 16 + hf * 8 + 4) = pk1;
      if (t == 127) {
        #pragma unroll
        for (int r = 0; r < 8; ++r)
          *(float*)(sm + HB + (nt * 2048 + (mt * 16 + r + 8 * hf) * 16 + p) * 4) = hv[r];
      }
    }
    if (t < 127) {
      unsigned u[4] = {(unsigned)xn.x, (unsigned)xn.y, (unsigned)xn.z, (unsigned)xn.w};
      #pragma unroll
      for (int j = 0; j < 16; ++j)
        sm[PAD + bs * 6912 + ((j >> 2) + 1) * 1152 + ((j & 3) + 1) * 192 + cs] =
            (char)((u[j >> 2] >> (8 * (j & 3))) & 0xffu);
    }
    __syncthreads();
  }

  for (int task = tid; task < 512; task += 256) {
    int bl = task >> 7, oc = task & 127;
    float acc = cls1_b[oc];
    const float4* w = (const float4*)(cls1_w + (size_t)oc * 2048);
    const float4* f = (const float4*)(sm + HB + bl * 8192);
    for (int kk = 0; kk < 512; ++kk) {
      float4 wv = w[kk], fv = f[kk];
      acc += wv.x * fv.x + wv.y * fv.y + wv.z * fv.z + wv.w * fv.w;
    }
    *(float*)(sm + HID + (bl * 128 + oc) * 4) = fmaxf(acc, 0.f);
  }
  __syncthreads();
  if (tid < 8) {
    int bl = tid >> 1, oc = tid & 1;
    float a = cls2_b[oc];
    const float* hw = cls2_w + oc * 128;
    const float* hh = (const float*)(sm + HID + bl * 512);
    for (int kk = 0; kk < 128; ++kk) a += hw[kk] * hh[kk];
    out[(blockIdx.x * 4 + bl) * 2 + oc] = a;
  }
}

extern "C" void kernel_launch(void* const* d_in, const int* in_sizes, int n_in,
                              void* d_out, int out_size, void* d_ws, size_t ws_size,
                              hipStream_t stream) {
  const float* x    = (const float*)d_in[0];
  const float* c1w  = (const float*)d_in[1];
  const float* c1b  = (const float*)d_in[2];
  const float* bg   = (const float*)d_in[3];
  const float* bb   = (const float*)d_in[4];
  const float* bm   = (const float*)d_in[5];
  const float* bv   = (const float*)d_in[6];
  const float* lw   = (const float*)d_in[7];
  const float* lb   = (const float*)d_in[8];
  const float* cw   = (const float*)d_in[9];
  const float* cb   = (const float*)d_in[10];
  const float* w1c  = (const float*)d_in[11];
  const float* b1c  = (const float*)d_in[12];
  const float* w2c  = (const float*)d_in[13];
  const float* b2c  = (const float*)d_in[14];
  float* out = (float*)d_out;
  char*  ws  = (char*)d_ws;

  _Float16* w1f = (_Float16*)(ws + WS_W1F);
  _Float16* w2f = (_Float16*)(ws + WS_W2F);
  float*    bcp = (float*)(ws + WS_BCP);
  unsigned char* wfrag = (unsigned char*)(ws + WS_WFRAG);
  unsigned char* xtb   = (unsigned char*)(ws + WS_XT);

  prep_enc1<<<24, 256, 0, stream>>>(c1w, c1b, bg, bb, bm, bv, w1f, bcp);
  prep_enc2<<<256, 256, 0, stream>>>(lw, w2f);
  prep_cell<<<896, 256, 0, stream>>>(cw, wfrag);
  enc_kernel<<<256, 256, 188672, stream>>>(x, w1f, w2f, bcp, lb, xtb);
  scan_kernel<<<512, 256, 259328, stream>>>(xtb, wfrag, cb, w1c, b1c, w2c, b2c, out);
}
